// smartorn_layer_82592221102095
// MI455X (gfx1250) — matrix operand order repaired and checked by reading, not compiled, not run on hardware
//
#include <hip/hip_runtime.h>

typedef __attribute__((ext_vector_type(16))) _Float16 v16h;
typedef __attribute__((ext_vector_type(8)))  _Float16 v8h;
typedef __attribute__((ext_vector_type(8)))  float    v8f;

#define NN_     4096
#define BATCH_  8
#define CHUNK_  256   // i-values staged per block iteration
#define TILEJ_  16
#define WAVES_  8

// strength(v) = (exp(v) - e^-1) / (e^1 - e^-1)
#define EM1_  0.36787944117144233f
#define ISC_  0.42545906059316383f
#define EPS_  1e-5f

// Load 8 consecutive floats from 16B-aligned shared memory via two b128 ds loads.
#define LOAD8F(dst, src, off)                                  \
  do {                                                         \
    const float4 _a = *(const float4*)&(src)[(off)];           \
    const float4 _b = *(const float4*)&(src)[(off) + 4];       \
    (dst)[0] = _a.x; (dst)[1] = _a.y; (dst)[2] = _a.z; (dst)[3] = _a.w; \
    (dst)[4] = _b.x; (dst)[5] = _b.y; (dst)[6] = _b.z; (dst)[7] = _b.w; \
  } while (0)

__global__ __launch_bounds__(256) void smartorn_fused_wmma(
    const float* __restrict__ x,        // (8, 4096)
    const float* __restrict__ position, // (4096, 3)
    const float* __restrict__ indir,    // (4096, 3)
    const float* __restrict__ outdir,   // (4096, 3)
    const float* __restrict__ power,    // (4096)
    const float* __restrict__ bias,     // (4096)
    float* __restrict__ out)            // (8, 4096)
{
  __shared__ __align__(16) float    spx[CHUNK_], spy[CHUNK_], spz[CHUNK_];
  __shared__ __align__(16) float    sox[CHUNK_], soy[CHUNK_], soz[CHUNK_];
  __shared__ __align__(16) _Float16 sxh[BATCH_ * CHUNK_];
  __shared__ __align__(16) float    sred[WAVES_ * 32 * 8];

  const int tid  = threadIdx.x;
  const int wave = tid >> 5;
  const int lane = tid & 31;
  const int h    = lane >> 4;   // which half of the wave
  const int nn   = lane & 15;   // column within j-tile (B/C/D N-index, A M-index)
  const int j    = blockIdx.x * TILEJ_ + nn;

  // Receiver-side (column j) quantities: loop-invariant, hoisted.
  const float pjx = position[j * 3 + 0];
  const float pjy = position[j * 3 + 1];
  const float pjz = position[j * 3 + 2];
  const float ijx = indir[j * 3 + 0];
  const float ijy = indir[j * 3 + 1];
  const float ijz = indir[j * 3 + 2];

  v8f c;
  #pragma unroll
  for (int r = 0; r < 8; ++r) c[r] = 0.0f;

  for (int it = 0; it < NN_ / CHUNK_; ++it) {
    const int ibase = it * CHUNK_;

    // ---- Stage sender chunk: position / outdir (SoA) and x (f16) ----
    {
      const int i = ibase + tid;
      spx[tid] = position[i * 3 + 0];
      spy[tid] = position[i * 3 + 1];
      spz[tid] = position[i * 3 + 2];
      sox[tid] = outdir[i * 3 + 0];
      soy[tid] = outdir[i * 3 + 1];
      soz[tid] = outdir[i * 3 + 2];
      #pragma unroll
      for (int b = 0; b < BATCH_; ++b)
        sxh[b * CHUNK_ + tid] = (_Float16)x[b * NN_ + i];
    }
    __syncthreads();

    // ---- A fragment: x tile 16(M=batch, rows 8..15 zero) x 32(K=i) ----
    // A layout (16-bit 16x32): elem idx -> K = h*16 + idx
    v16h a;
    if (nn < BATCH_) {
      const _Float16* row = &sxh[nn * CHUNK_ + wave * 32 + h * 16];
      const v8h lo = *(const v8h*)(row);        // K = h*16 + 0..7
      const v8h hi = *(const v8h*)(row + 8);    // K = h*16 + 8..15
      #pragma unroll
      for (int k = 0; k < 8; ++k) { a[k] = lo[k]; a[8 + k] = hi[k]; }
    } else {
      #pragma unroll
      for (int k = 0; k < 16; ++k) a[k] = (_Float16)0.0f;
    }

    // ---- B fragment: gate tile 32(K=i) x 16(N=j), computed in registers ----
    // B layout (16-bit 32x16): elem idx -> K = h*16 + idx, column = lane&15
    const int kb = wave * 32 + h * 16;  // local sender index base for this lane
    float PX[16], PY[16], PZ[16], OX[16], OY[16], OZ[16];
    LOAD8F(PX, spx, kb); LOAD8F(&PX[8], spx, kb + 8);
    LOAD8F(PY, spy, kb); LOAD8F(&PY[8], spy, kb + 8);
    LOAD8F(PZ, spz, kb); LOAD8F(&PZ[8], spz, kb + 8);
    LOAD8F(OX, sox, kb); LOAD8F(&OX[8], sox, kb + 8);
    LOAD8F(OY, soy, kb); LOAD8F(&OY[8], soy, kb + 8);
    LOAD8F(OZ, soz, kb); LOAD8F(&OZ[8], soz, kb + 8);

    v16h bf;
    #pragma unroll
    for (int k = 0; k < 16; ++k) {
      // delta_p = pos[j] - pos[i]   (i = sender = kb+k, j = receiver = this column)
      const float dx = pjx - PX[k];
      const float dy = pjy - PY[k];
      const float dz = pjz - PZ[k];
      const float d2 = fmaxf(dx * dx + dy * dy + dz * dz, EPS_);
      const float rn = __frsqrt_rn(d2);                       // 1 / ||delta_p||
      const float da = (dx * OX[k] + dy * OY[k] + dz * OZ[k]) * rn;  // . outdir_i
      const float db = (dx * ijx + dy * ijy + dz * ijz) * rn;        // . indir_j
      const float alpha = (__expf(da) - EM1_) * ISC_;
      const float beta  = (__expf(db) - EM1_) * ISC_;
      bf[k] = (_Float16)(alpha * beta);
    }

    // ---- Accumulate: C += A(16x32) * B(32x16) ----
    c = __builtin_amdgcn_wmma_f32_16x16x32_f16(
        /*neg_a=*/false, a, /*neg_b=*/false, bf,
        /*c_mod=*/(short)0, c, /*reuse_a=*/false, /*reuse_b=*/false);

    __syncthreads();  // protect staging buffers before next iteration
  }

  // ---- Cross-wave reduction of partial accumulators ----
  #pragma unroll
  for (int r = 0; r < 8; ++r) sred[(wave * 32 + lane) * 8 + r] = c[r];
  __syncthreads();

  if (wave == 0) {
    float acc[8];
    #pragma unroll
    for (int r = 0; r < 8; ++r) {
      float s = 0.0f;
      #pragma unroll
      for (int w2 = 0; w2 < WAVES_; ++w2) s += sred[(w2 * 32 + lane) * 8 + r];
      acc[r] = s;
    }
    // C layout: lanes 0..15 hold M = r (= batch 0..7) for VGPR r, N = lane.
    if (h == 0) {
      const float pw = power[j];
      const float bs = bias[j];
      #pragma unroll
      for (int r = 0; r < 8; ++r) {
        const float s = acc[r] * pw - bs;
        out[r * NN_ + j] = (s > 0.0f) ? s : (__expf(s) - 1.0f);  // elu
      }
    }
  }
}

extern "C" void kernel_launch(void* const* d_in, const int* in_sizes, int n_in,
                              void* d_out, int out_size, void* d_ws, size_t ws_size,
                              hipStream_t stream) {
  const float* x        = (const float*)d_in[0];
  const float* position = (const float*)d_in[1];
  const float* indir    = (const float*)d_in[2];
  const float* outdir   = (const float*)d_in[3];
  const float* power    = (const float*)d_in[4];
  const float* bias     = (const float*)d_in[5];
  float* out = (float*)d_out;

  dim3 grid(NN_ / TILEJ_);   // 256 blocks: one 16-column j-tile each
  dim3 block(32 * WAVES_);   // 8 waves (wave32)
  smartorn_fused_wmma<<<grid, block, 0, stream>>>(x, position, indir, outdir,
                                                  power, bias, out);
}
